// MFN_64750926955038
// MI455X (gfx1250) — matrix operand order repaired and checked by reading, not compiled, not run on hardware
//
#include <hip/hip_runtime.h>
#include <hip/hip_bf16.h>

// MFN on MI455X (gfx1250): single persistent kernel, wave32, bf16 WMMA GEMMs
// (v_wmma_f32_16x16x32_bf16) with f32 accumulate/state. Grid-wide software
// barrier between dependent phases replaces ~520 per-step kernel launches.
// Register blocking: 2x2 output macro-tiles (4 WMMA / k-step, 32B loaded per
// WMMA) for the MLP GEMMs; shared-A gate-quad (4 WMMA / k-step) for the LSTM.

typedef __bf16 bf16;
typedef __attribute__((ext_vector_type(16))) __bf16 v16bf;
typedef __attribute__((ext_vector_type(8)))  __bf16 v8bf;
typedef __attribute__((ext_vector_type(8)))  float  v8f;

#define NBLK  64
#define NTHR  256
#define NWAVE 512          // 64 blocks * 8 wave32 waves
#define NTOT  16384        // total threads

#define T_STEPS 64
#define NROW 512           // batch
#define XW   556           // raw x row width (300+128+128)
#define XSW  576           // padded bf16 x row: [l:320][a:128][v:128]

__device__ __forceinline__ bf16 f2bf(float f) {
  unsigned u = __float_as_uint(f);
  u += 0x7FFFu + ((u >> 16) & 1u);          // round-to-nearest-even
  unsigned short h = (unsigned short)(u >> 16);
  return __builtin_bit_cast(bf16, h);
}
__device__ __forceinline__ float sigf(float x) { return 1.0f / (1.0f + __expf(-x)); }
__device__ __forceinline__ v8f vzero() {
  v8f c;
#pragma unroll
  for (int i = 0; i < 8; ++i) c[i] = 0.f;
  return c;
}

// A-fragment: two aligned 16B loads, lanes 0-15 row m (K 0..15),
// lanes 16-31 same rows (K 16..31), matching the contiguous B fill.
__device__ __forceinline__ v16bf ldA16(const bf16* p) {
  v8bf a0 = *(const v8bf*)p;
  v8bf a1 = *(const v8bf*)(p + 8);
  return __builtin_shufflevector(a0, a1, 0,1,2,3,4,5,6,7,8,9,10,11,12,13,14,15);
}
#define WMMA_BF16(a, b, c) \
  __builtin_amdgcn_wmma_f32_16x16x32_bf16(false, (a), false, (b), (short)0, (c), false, false)

struct P {
  unsigned* arrive; unsigned* gen;
  bf16* hb[2]; float* cf[2]; float* memf; float* hf;
  const bf16 *wih_l, *wih_a, *wih_v, *whh_l, *whh_a, *whh_v;
  const bf16 *a1w1, *a1w2, *a2w1, *a2w2, *g1w1, *g1w2, *g2w1, *g2w2;
  bf16* xs; bf16* cstarb; bf16* z1b; float* lg; bf16* bothb;
  bf16* z2b; bf16* zg1b; bf16* zg2b;
  float* chatf; float* g1f; float* g2f; float* logf; float* lasths; float* pmax;
  const float* x;
  const float *bih_l, *bhh_l, *bih_a, *bhh_a, *bih_v, *bhh_v;
  const float *a1b1, *a1b2, *a2b1, *a2b2, *g1b1, *g1b2, *g2b1, *g2b2;
  const float *flw, *flb, *faw, *fab, *fvw, *fvb, *ow1, *ob1, *ow2, *ob2;
  float* out;
};

// Grid-wide barrier: generation counter in L2, s_sleep spin (64 co-resident blocks).
__device__ __forceinline__ void gsync(unsigned* arrive, unsigned* gen) {
  __syncthreads();
  if (threadIdx.x == 0) {
    volatile unsigned* vgen = gen;
    unsigned g = *vgen;
    __threadfence();
    unsigned a = atomicAdd(arrive, 1u);
    if (a == (unsigned)(NBLK - 1)) {
      atomicExch(arrive, 0u);
      __threadfence();
      atomicAdd(gen, 1u);
    } else {
      while (*vgen == g) __builtin_amdgcn_s_sleep(1);
    }
    __threadfence();
  }
  __syncthreads();
}

// 2x2 macro-tile (32x32) of Z = A(row-major,lda) @ W^T (W out x in, row-major,ldw).
// Per k-step: 2 A-frags + 2 B-frags -> 4 WMMAs (c[0]:(m0,n0) c[1]:(m0,n0+16)
// c[2]:(m0+16,n0) c[3]:(m0+16,n0+16)).
__device__ __forceinline__ void tile_mm2x2(const bf16* A, int lda, const bf16* W, int ldw,
                                           int m0, int n0, int K, v8f c[4]) {
  int lane = threadIdx.x & 31;
  int lo = lane & 15, hi = lane >> 4;
  const bf16* ap0 = A + (size_t)(m0 + lo) * lda + hi * 16;
  const bf16* ap1 = ap0 + (size_t)16 * lda;
  const bf16* wp0 = W + (size_t)(n0 + lo) * ldw + hi * 16;
  const bf16* wp1 = wp0 + (size_t)16 * ldw;
  for (int k = 0; k < K; k += 32) {
    v16bf a0 = ldA16(ap0 + k);
    v16bf a1 = ldA16(ap1 + k);
    v16bf b0 = *(const v16bf*)(wp0 + k);
    v16bf b1 = *(const v16bf*)(wp1 + k);
    c[0] = WMMA_BF16(a0, b0, c[0]);
    c[1] = WMMA_BF16(a0, b1, c[1]);
    c[2] = WMMA_BF16(a1, b0, c[2]);
    c[3] = WMMA_BF16(a1, b1, c[3]);
  }
}

// Gate-quad: one A stream shared by 4 output-column tiles at n0 + g*nstep
// (LSTM gates i,f,g,o). Per k-step: 1 A-frag + 4 B-frags -> 4 WMMAs.
__device__ __forceinline__ void tile_mm_quad(const bf16* A, int lda, const bf16* W, int ldw,
                                             int m0, int n0, int nstep, int K, v8f c[4]) {
  int lane = threadIdx.x & 31;
  int lo = lane & 15, hi = lane >> 4;
  const bf16* ap  = A + (size_t)(m0 + lo) * lda + hi * 16;
  const bf16* wp0 = W + (size_t)(n0 + 0 * nstep + lo) * ldw + hi * 16;
  const bf16* wp1 = W + (size_t)(n0 + 1 * nstep + lo) * ldw + hi * 16;
  const bf16* wp2 = W + (size_t)(n0 + 2 * nstep + lo) * ldw + hi * 16;
  const bf16* wp3 = W + (size_t)(n0 + 3 * nstep + lo) * ldw + hi * 16;
  for (int k = 0; k < K; k += 32) {
    v16bf a  = ldA16(ap + k);
    v16bf b0 = *(const v16bf*)(wp0 + k);
    v16bf b1 = *(const v16bf*)(wp1 + k);
    v16bf b2 = *(const v16bf*)(wp2 + k);
    v16bf b3 = *(const v16bf*)(wp3 + k);
    c[0] = WMMA_BF16(a, b0, c[0]);
    c[1] = WMMA_BF16(a, b1, c[1]);
    c[2] = WMMA_BF16(a, b2, c[2]);
    c[3] = WMMA_BF16(a, b3, c[3]);
  }
}

__global__ void conv_bf16(const float* __restrict__ src, bf16* __restrict__ dst,
                          int rows, int kin, int kout) {
  int total = rows * kout;
  for (int i = blockIdx.x * blockDim.x + threadIdx.x; i < total;
       i += gridDim.x * blockDim.x) {
    int r = i / kout, c = i - r * kout;
    dst[i] = (c < kin) ? f2bf(src[(size_t)r * kin + c])
                       : __builtin_bit_cast(bf16, (unsigned short)0);
  }
}

__global__ __launch_bounds__(NTHR) void mfn_main(P p) {
  const int gt   = blockIdx.x * NTHR + threadIdx.x;
  const int gw   = (blockIdx.x << 3) + (threadIdx.x >> 5);
  const int lane = threadIdx.x & 31;
  const int lo   = lane & 15, hi = lane >> 4;

  for (int t = 0; t < T_STEPS; ++t) {
    const bf16*  hbR = p.hb[t & 1];
    bf16*        hbW = p.hb[(t + 1) & 1];
    const float* cfR = p.cf[t & 1];
    float*       cfW = p.cf[(t + 1) & 1];
    const float* xt  = p.x + (size_t)t * NROW * XW;

    // ---------------- phase 0: x_t -> padded bf16; c_old -> cStar[:, :512]
    for (int i = gt; i < NROW * XSW; i += NTOT) {
      int n = i / XSW, c = i - n * XSW;
      float v = 0.f;
      if (c < 300)       v = xt[(size_t)n * XW + c];
      else if (c >= 320) v = xt[(size_t)n * XW + (c - 20)];  // a:[320,448)->[300,428), v:[448,576)->[428,556)
      p.xs[i] = f2bf(v);
    }
    for (int i = gt; i < NROW * 512; i += NTOT) {
      int n = i >> 9, c = i & 511;
      p.cstarb[(size_t)n * 1024 + c] = f2bf(cfR[i]);
    }
    gsync(p.arrive, p.gen);

    // ---------------- phase 1: fused 3x LSTM (gate-quad GEMMs + cell update)
    for (int tile = gw; tile < 1024; tile += NWAVE) {
      int mt = tile >> 5, ct = tile & 31;
      int H, hoff, xoff, Kx, lc;
      const bf16 *Wih, *Whh; const float *bih, *bhh;
      if (ct < 16)      { H=256; hoff=0;   xoff=0;   Kx=320; lc=ct;    Wih=p.wih_l; Whh=p.whh_l; bih=p.bih_l; bhh=p.bhh_l; }
      else if (ct < 24) { H=128; hoff=256; xoff=320; Kx=128; lc=ct-16; Wih=p.wih_a; Whh=p.whh_a; bih=p.bih_a; bhh=p.bhh_a; }
      else              { H=128; hoff=384; xoff=448; Kx=128; lc=ct-24; Wih=p.wih_v; Whh=p.whh_v; bih=p.bih_v; bhh=p.bhh_v; }
      int m0 = mt * 16;
      v8f acc[4] = {vzero(), vzero(), vzero(), vzero()};
      tile_mm_quad(p.xs + xoff, XSW, Wih, Kx, m0, lc * 16, H, Kx, acc);
      tile_mm_quad(hbR + hoff, 512, Whh, H,  m0, lc * 16, H, H,  acc);
      int col = lc * 16 + lo;
      float b0 = bih[0*H+col] + bhh[0*H+col];
      float b1 = bih[1*H+col] + bhh[1*H+col];
      float b2 = bih[2*H+col] + bhh[2*H+col];
      float b3 = bih[3*H+col] + bhh[3*H+col];
#pragma unroll
      for (int r = 0; r < 8; ++r) {
        int row = m0 + r + hi * 8;
        float iv = sigf(acc[0][r] + b0);
        float fv = sigf(acc[1][r] + b1);
        float gv = tanhf(acc[2][r] + b2);
        float ov = sigf(acc[3][r] + b3);
        float cold = cfR[row * 512 + hoff + col];
        float c2 = fv * cold + iv * gv;
        float h2 = ov * tanhf(c2);
        cfW[row * 512 + hoff + col] = c2;
        p.hf[row * 512 + hoff + col] = h2;
        hbW[row * 512 + hoff + col] = f2bf(h2);
        p.cstarb[(size_t)row * 1024 + 512 + hoff + col] = f2bf(c2);
      }
    }
    gsync(p.arrive, p.gen);

    // ---------------- phase 2: z1 = relu(cStar @ att1_w1^T + b1)  (512x512)
    for (int tile = gw; tile < 256; tile += NWAVE) {       // 16x16 macro-tiles of 32x32
      int m0 = (tile >> 4) * 32, n0 = (tile & 15) * 32;
      v8f c[4] = {vzero(), vzero(), vzero(), vzero()};
      tile_mm2x2(p.cstarb, 1024, p.a1w1, 1024, m0, n0, 1024, c);
#pragma unroll
      for (int q = 0; q < 4; ++q) {
        int mq = m0 + (q >> 1) * 16, ncol = n0 + (q & 1) * 16 + lo;
        float b = p.a1b1[ncol];
#pragma unroll
        for (int r = 0; r < 8; ++r) {
          float v = c[q][r] + b; v = v > 0.f ? v : 0.f;
          p.z1b[(size_t)(mq + r + hi * 8) * 512 + ncol] = f2bf(v);
        }
      }
    }
    gsync(p.arrive, p.gen);

    // ---------------- phase 3: logits = z1 @ att1_w2^T + b2  (512x1024, f32)
    for (int tile = gw; tile < 512; tile += NWAVE) {       // 16x32 macro-tiles
      int m0 = (tile >> 5) * 32, n0 = (tile & 31) * 32;
      v8f c[4] = {vzero(), vzero(), vzero(), vzero()};
      tile_mm2x2(p.z1b, 512, p.a1w2, 512, m0, n0, 512, c);
#pragma unroll
      for (int q = 0; q < 4; ++q) {
        int mq = m0 + (q >> 1) * 16, ncol = n0 + (q & 1) * 16 + lo;
        float b = p.a1b2[ncol];
#pragma unroll
        for (int r = 0; r < 8; ++r)
          p.lg[(size_t)(mq + r + hi * 8) * 1024 + ncol] = c[q][r] + b;
      }
    }
    gsync(p.arrive, p.gen);

    // ---------------- phase 4: row softmax, attended = att*cStar, mem mirror
    {
      int row = gw;                       // exactly 512 waves, one row each
      const float* lgr = p.lg + (size_t)row * 1024;
      float mx = -3.4e38f;
      for (int j = lane; j < 1024; j += 32) mx = fmaxf(mx, lgr[j]);
      for (int o = 16; o; o >>= 1) mx = fmaxf(mx, __shfl_xor(mx, o, 32));
      float sum = 0.f;
      for (int j = lane; j < 1024; j += 32) sum += __expf(lgr[j] - mx);
      for (int o = 16; o; o >>= 1) sum += __shfl_xor(sum, o, 32);
      float inv = 1.f / sum;
      for (int j = lane; j < 1024; j += 32) {
        float att = __expf(lgr[j] - mx) * inv;
        float cs = (j < 512) ? cfR[row * 512 + j] : cfW[row * 512 + (j - 512)];
        p.bothb[(size_t)row * 1280 + j] = f2bf(att * cs);
      }
      for (int j = lane; j < 256; j += 32)
        p.bothb[(size_t)row * 1280 + 1024 + j] = f2bf(p.memf[row * 256 + j]);
    }
    gsync(p.arrive, p.gen);

    // ---------------- phase 5: z2 | zg1 | zg2  (three 512x512 relu GEMMs)
    for (int id = gw; id < 768; id += NWAVE) {             // 3 * 256 macro-tiles
      int which = id >> 8;
      int t5 = id & 255;
      int m0 = (t5 >> 4) * 32, n0 = (t5 & 15) * 32;
      v8f c[4] = {vzero(), vzero(), vzero(), vzero()};
      const float* bias; bf16* dst;
      if (which == 0)      { tile_mm2x2(p.bothb, 1280, p.a2w1, 1024, m0, n0, 1024, c); bias = p.a2b1; dst = p.z2b; }
      else if (which == 1) { tile_mm2x2(p.bothb, 1280, p.g1w1, 1280, m0, n0, 1280, c); bias = p.g1b1; dst = p.zg1b; }
      else                 { tile_mm2x2(p.bothb, 1280, p.g2w1, 1280, m0, n0, 1280, c); bias = p.g2b1; dst = p.zg2b; }
#pragma unroll
      for (int q = 0; q < 4; ++q) {
        int mq = m0 + (q >> 1) * 16, ncol = n0 + (q & 1) * 16 + lo;
        float b = bias[ncol];
#pragma unroll
        for (int r = 0; r < 8; ++r) {
          float v = c[q][r] + b; v = v > 0.f ? v : 0.f;
          dst[(size_t)(mq + r + hi * 8) * 512 + ncol] = f2bf(v);
        }
      }
    }
    gsync(p.arrive, p.gen);

    // ---------------- phase 6: cHat=tanh | g1=sig | g2=sig  (three 512x256)
    for (int id = gw; id < 384; id += NWAVE) {             // 3 * 128 macro-tiles
      int which = id >> 7;
      int t6 = id & 127;
      int m0 = (t6 >> 3) * 32, n0 = (t6 & 7) * 32;
      v8f c[4] = {vzero(), vzero(), vzero(), vzero()};
      const float* bias; float* dst;
      if (which == 0)      { tile_mm2x2(p.z2b,  512, p.a2w2, 512, m0, n0, 512, c); bias = p.a2b2; dst = p.chatf; }
      else if (which == 1) { tile_mm2x2(p.zg1b, 512, p.g1w2, 512, m0, n0, 512, c); bias = p.g1b2; dst = p.g1f; }
      else                 { tile_mm2x2(p.zg2b, 512, p.g2w2, 512, m0, n0, 512, c); bias = p.g2b2; dst = p.g2f; }
#pragma unroll
      for (int q = 0; q < 4; ++q) {
        int mq = m0 + (q >> 1) * 16, ncol = n0 + (q & 1) * 16 + lo;
        float b = bias[ncol];
#pragma unroll
        for (int r = 0; r < 8; ++r) {
          float v = c[q][r] + b;
          v = (which == 0) ? tanhf(v) : sigf(v);
          dst[(size_t)(mq + r + hi * 8) * 256 + ncol] = v;
        }
      }
    }
    gsync(p.arrive, p.gen);

    // ---------------- phase 7: mem = g1*mem + g2*cHat
    for (int i = gt; i < 512 * 256; i += NTOT)
      p.memf[i] = p.g1f[i] * p.memf[i] + p.g2f[i] * p.chatf[i];
    gsync(p.arrive, p.gen);
  }

  // ================= final head (tiny FLOPs, scalar f32) =================
  // F1: per-modality logits (512 x 384 = [l:128 | a:128 | v:128])
  for (int idx = gt; idx < 512 * 384; idx += NTOT) {
    int n = idx / 384, j = idx - n * 384;
    int m = j >> 7, jj = j & 127;
    const float *w, *b; int hoff, Hm;
    if (m == 0)      { w = p.flw; b = p.flb; hoff = 0;   Hm = 256; }
    else if (m == 1) { w = p.faw; b = p.fab; hoff = 256; Hm = 128; }
    else             { w = p.fvw; b = p.fvb; hoff = 384; Hm = 128; }
    float s = b[jj];
    const float* hrow = p.hf + (size_t)n * 512 + hoff;
    const float* wrow = w + (size_t)jj * Hm;
    for (int k = 0; k < Hm; ++k) s += hrow[k] * wrow[k];
    p.logf[idx] = s;
  }
  gsync(p.arrive, p.gen);

  // F2: per-block max per modality
  {
    float m3[3] = {-3.4e38f, -3.4e38f, -3.4e38f};
    for (int idx = gt; idx < 512 * 384; idx += NTOT) {
      int mod = (idx % 384) >> 7;
      m3[mod] = fmaxf(m3[mod], p.logf[idx]);
    }
    __shared__ float red[3][8];
#pragma unroll
    for (int m = 0; m < 3; ++m) {
      float v = m3[m];
      for (int o = 16; o; o >>= 1) v = fmaxf(v, __shfl_xor(v, o, 32));
      if (lane == 0) red[m][threadIdx.x >> 5] = v;
    }
    __syncthreads();
    if (threadIdx.x < 3) {
      float v = red[threadIdx.x][0];
      for (int w2 = 1; w2 < 8; ++w2) v = fmaxf(v, red[threadIdx.x][w2]);
      p.pmax[blockIdx.x * 3 + threadIdx.x] = v;
    }
  }
  gsync(p.arrive, p.gen);

  // F3: lav = sum_m p*log(p) with p=exp(logit-gmax); last_hs = [lav | mem]
  {
    float gm[3] = {-3.4e38f, -3.4e38f, -3.4e38f};
    for (int b2 = 0; b2 < NBLK; ++b2) {
#pragma unroll
      for (int m = 0; m < 3; ++m) gm[m] = fmaxf(gm[m], p.pmax[b2 * 3 + m]);
    }
    for (int idx = gt; idx < 512 * 384; idx += NTOT) {
      int n = idx / 384, j = idx - n * 384;
      float v;
      if (j < 128) {
        v = 0.f;
#pragma unroll
        for (int m = 0; m < 3; ++m) {
          float tt = p.logf[(size_t)n * 384 + m * 128 + j] - gm[m];
          v += __expf(tt) * tt;     // p^1 * log(p)
        }
      } else {
        v = p.memf[(size_t)n * 256 + (j - 128)];
      }
      p.lasths[idx] = v;
    }
  }
  gsync(p.arrive, p.gen);

  // F4: out = relu(last_hs @ o_w1^T + b1) @ o_w2^T + b2   (wave per row)
  {
    int row = gw;
    const float* lh = p.lasths + (size_t)row * 384;
    float acc = 0.f;
    for (int j = lane; j < 512; j += 32) {
      float s = p.ob1[j];
      const float* wr = p.ow1 + (size_t)j * 384;
      for (int k = 0; k < 384; ++k) s += lh[k] * wr[k];
      s = s > 0.f ? s : 0.f;
      acc += s * p.ow2[j];
    }
    for (int o = 16; o; o >>= 1) acc += __shfl_xor(acc, o, 32);
    if (lane == 0) p.out[row] = acc + p.ob2[0];
  }
}

extern "C" void kernel_launch(void* const* d_in, const int* in_sizes, int n_in,
                              void* d_out, int out_size, void* d_ws, size_t ws_size,
                              hipStream_t stream) {
  (void)in_sizes; (void)n_in; (void)out_size;
  char* ws = (char*)d_ws;
  size_t off = 0;
  auto alloc = [&](size_t b) { size_t o = off; off = (off + b + 255) & ~(size_t)255; return o; };

  // zero-initialized region first (one memset covers it)
  size_t oSync = alloc(256);
  size_t oHb0  = alloc((size_t)512*512*2);
  size_t oHb1  = alloc((size_t)512*512*2);
  size_t oCf0  = alloc((size_t)512*512*4);
  size_t oCf1  = alloc((size_t)512*512*4);
  size_t oMem  = alloc((size_t)512*256*4);
  size_t oHf   = alloc((size_t)512*512*4);
  size_t zeroEnd = off;
  // bf16 weights
  size_t oWihL = alloc((size_t)1024*320*2);
  size_t oWihA = alloc((size_t)512*128*2);
  size_t oWihV = alloc((size_t)512*128*2);
  size_t oWhhL = alloc((size_t)1024*256*2);
  size_t oWhhA = alloc((size_t)512*128*2);
  size_t oWhhV = alloc((size_t)512*128*2);
  size_t oA1W1 = alloc((size_t)512*1024*2);
  size_t oA1W2 = alloc((size_t)1024*512*2);
  size_t oA2W1 = alloc((size_t)512*1024*2);
  size_t oA2W2 = alloc((size_t)256*512*2);
  size_t oG1W1 = alloc((size_t)512*1280*2);
  size_t oG1W2 = alloc((size_t)256*512*2);
  size_t oG2W1 = alloc((size_t)512*1280*2);
  size_t oG2W2 = alloc((size_t)256*512*2);
  // activations
  size_t oXs   = alloc((size_t)512*576*2);
  size_t oCst  = alloc((size_t)512*1024*2);
  size_t oZ1   = alloc((size_t)512*512*2);
  size_t oLg   = alloc((size_t)512*1024*4);
  size_t oBoth = alloc((size_t)512*1280*2);
  size_t oZ2   = alloc((size_t)512*512*2);
  size_t oZg1  = alloc((size_t)512*512*2);
  size_t oZg2  = alloc((size_t)512*512*2);
  size_t oChat = alloc((size_t)512*256*4);
  size_t oG1f  = alloc((size_t)512*256*4);
  size_t oG2f  = alloc((size_t)512*256*4);
  size_t oLogf = alloc((size_t)512*384*4);
  size_t oLast = alloc((size_t)512*384*4);
  size_t oPmax = alloc((size_t)64*3*4);
  if (ws_size < off) return;   // ~23 MB required

  auto conv = [&](int in_idx, size_t dstoff, int rows, int kin, int kout) {
    int total = rows * kout;
    int blocks = (total + 255) / 256; if (blocks > 2048) blocks = 2048;
    conv_bf16<<<blocks, 256, 0, stream>>>((const float*)d_in[in_idx],
                                          (bf16*)(ws + dstoff), rows, kin, kout);
  };
  conv(1,  oWihL, 1024, 300, 320);   // Wih_l (pad K 300->320)
  conv(2,  oWhhL, 1024, 256, 256);   // Whh_l
  conv(5,  oWihA, 512, 128, 128);    // Wih_a
  conv(6,  oWhhA, 512, 128, 128);    // Whh_a
  conv(9,  oWihV, 512, 128, 128);    // Wih_v
  conv(10, oWhhV, 512, 128, 128);    // Whh_v
  conv(13, oA1W1, 512, 1024, 1024);
  conv(15, oA1W2, 1024, 512, 512);
  conv(17, oA2W1, 512, 1024, 1024);
  conv(19, oA2W2, 256, 512, 512);
  conv(21, oG1W1, 512, 1280, 1280);
  conv(23, oG1W2, 256, 512, 512);
  conv(25, oG2W1, 512, 1280, 1280);
  conv(27, oG2W2, 256, 512, 512);

  (void)hipMemsetAsync(d_ws, 0, zeroEnd, stream);   // sync counters + initial states

  P p;
  p.arrive = (unsigned*)(ws + oSync);
  p.gen    = (unsigned*)(ws + oSync + 128);
  p.hb[0] = (bf16*)(ws + oHb0);  p.hb[1] = (bf16*)(ws + oHb1);
  p.cf[0] = (float*)(ws + oCf0); p.cf[1] = (float*)(ws + oCf1);
  p.memf = (float*)(ws + oMem);  p.hf = (float*)(ws + oHf);
  p.wih_l = (const bf16*)(ws + oWihL); p.wih_a = (const bf16*)(ws + oWihA);
  p.wih_v = (const bf16*)(ws + oWihV); p.whh_l = (const bf16*)(ws + oWhhL);
  p.whh_a = (const bf16*)(ws + oWhhA); p.whh_v = (const bf16*)(ws + oWhhV);
  p.a1w1 = (const bf16*)(ws + oA1W1); p.a1w2 = (const bf16*)(ws + oA1W2);
  p.a2w1 = (const bf16*)(ws + oA2W1); p.a2w2 = (const bf16*)(ws + oA2W2);
  p.g1w1 = (const bf16*)(ws + oG1W1); p.g1w2 = (const bf16*)(ws + oG1W2);
  p.g2w1 = (const bf16*)(ws + oG2W1); p.g2w2 = (const bf16*)(ws + oG2W2);
  p.xs = (bf16*)(ws + oXs); p.cstarb = (bf16*)(ws + oCst); p.z1b = (bf16*)(ws + oZ1);
  p.lg = (float*)(ws + oLg); p.bothb = (bf16*)(ws + oBoth);
  p.z2b = (bf16*)(ws + oZ2); p.zg1b = (bf16*)(ws + oZg1); p.zg2b = (bf16*)(ws + oZg2);
  p.chatf = (float*)(ws + oChat); p.g1f = (float*)(ws + oG1f); p.g2f = (float*)(ws + oG2f);
  p.logf = (float*)(ws + oLogf); p.lasths = (float*)(ws + oLast); p.pmax = (float*)(ws + oPmax);
  p.x = (const float*)d_in[0];
  p.bih_l = (const float*)d_in[3];  p.bhh_l = (const float*)d_in[4];
  p.bih_a = (const float*)d_in[7];  p.bhh_a = (const float*)d_in[8];
  p.bih_v = (const float*)d_in[11]; p.bhh_v = (const float*)d_in[12];
  p.a1b1 = (const float*)d_in[14]; p.a1b2 = (const float*)d_in[16];
  p.a2b1 = (const float*)d_in[18]; p.a2b2 = (const float*)d_in[20];
  p.g1b1 = (const float*)d_in[22]; p.g1b2 = (const float*)d_in[24];
  p.g2b1 = (const float*)d_in[26]; p.g2b2 = (const float*)d_in[28];
  p.flw = (const float*)d_in[29]; p.flb = (const float*)d_in[30];
  p.faw = (const float*)d_in[31]; p.fab = (const float*)d_in[32];
  p.fvw = (const float*)d_in[33]; p.fvb = (const float*)d_in[34];
  p.ow1 = (const float*)d_in[35]; p.ob1 = (const float*)d_in[36];
  p.ow2 = (const float*)d_in[37]; p.ob2 = (const float*)d_in[38];
  p.out = (float*)d_out;

  mfn_main<<<NBLK, NTHR, 0, stream>>>(p);
}
